// DeepNoSAF_32280974197076
// MI455X (gfx1250) — compile-verified, matrix operand order checked by reading, not run on hardware
//
#include <hip/hip_runtime.h>
#include <hip/hip_bf16.h>
#include <cstddef>
#include <cstdint>

// ---------------------------------------------------------------------------
// CDNA5 (gfx1250) wave32 WMMA helpers
// ---------------------------------------------------------------------------
typedef __attribute__((ext_vector_type(16))) _Float16 v16h;
typedef __attribute__((ext_vector_type(8)))  _Float16 half8;
typedef __attribute__((ext_vector_type(8)))  float    v8f;

__device__ __forceinline__ v8f wmma_f16(v16h a, v16h b, v8f c) {
  // D = A(16x32 f16) * B(32x16 f16) + C(16x16 f32)
  return __builtin_amdgcn_wmma_f32_16x16x32_f16(
      /*neg_a=*/false, a, /*neg_b=*/false, b,
      /*c_mod=*/(short)0, c, /*reuse_a=*/false, /*reuse_b=*/false);
}

// Load one 16x32 f16 fragment from a row-major [16][ld] tile (A) or from a
// transposed weight matrix Wt laid out [n][K] row-major (B).  Per the CDNA5
// 16-bit A layout: lanes 0-15 hold row (lane&15), K = k0+{0..7, 16..23};
// lanes 16-31 hold the same row with K = k0+{8..15, 24..31}.  ld must be a
// multiple of 8 halfs so the two half8 loads are 16B aligned.
__device__ __forceinline__ v16h load_frag16(const _Float16* base, int ld,
                                            int k0, int lane) {
  const int row = lane & 15;
  const int off = (lane >> 4) << 3;  // 0 or 8
  const _Float16* p = base + row * ld + k0 + off;
  half8 lo = *reinterpret_cast<const half8*>(p);
  half8 hi = *reinterpret_cast<const half8*>(p + 16);
  v16h a;
#pragma unroll
  for (int i = 0; i < 8; ++i) { a[i] = lo[i]; a[i + 8] = hi[i]; }
  return a;
}

// ---------------------------------------------------------------------------
// Async global->LDS copy (CDNA5 GLOBAL_LOAD_ASYNC_TO_LDS, ASYNCcnt-tracked).
// The low 32 bits of a generic pointer to __shared__ memory are the
// wave-relative LDS byte address (ISA 10.2), which is what VDST expects.
// ---------------------------------------------------------------------------
__device__ __forceinline__ void async_copy_b128(const void* gptr, void* lptr) {
  uint32_t laddr = (uint32_t)(uintptr_t)lptr;
  asm volatile("global_load_async_to_lds_b128 %0, %1, off"
               :: "v"(laddr), "v"(gptr)
               : "memory");
}
__device__ __forceinline__ void wait_async0() {
  asm volatile("s_wait_asynccnt 0" ::: "memory");
}

// ---------------------------------------------------------------------------
// Constants
// ---------------------------------------------------------------------------
#define FDIM 128
#define HIDD 64
#define LDA  136   // 128 + 8 halfs pad  (rows stay 16B aligned, kills 64-bank conflicts)
#define LDA2 72    // 64 + 8 halfs pad
#define LDAE 40    // 32 + 8 halfs pad (encoder K=16 padded to 32)

// ---------------------------------------------------------------------------
// NodeWeightLearner: nw = sigmoid(leakyrelu(z@W1+b1, 0.2) @ W2 + b2)
// z is the f16 A tile ldsZ[16][LDA]; result f32 in ldsNW[16][128].
// Block = 256 threads = 8 wave32.
// ---------------------------------------------------------------------------
__device__ __forceinline__ void learner_block(
    const _Float16* ldsZ, _Float16* ldsZ2, float* ldsNW,
    const _Float16* Wt1, const float* b1,
    const _Float16* Wt2, const float* b2, int lane, int wave) {
  // GEMM 16x128 @ 128x64 -> 16x64 on waves 0..3
  if (wave < 4) {
    v8f acc = {};
#pragma unroll
    for (int k0 = 0; k0 < FDIM; k0 += 32) {
      v16h a = load_frag16(ldsZ, LDA, k0, lane);
      v16h b = load_frag16(Wt1 + wave * 16 * FDIM, FDIM, k0, lane);
      acc = wmma_f16(a, b, acc);
    }
    int n = wave * 16 + (lane & 15);
    int mb = (lane >> 4) << 3;
    float bias = b1[n];
#pragma unroll
    for (int r = 0; r < 8; ++r) {
      float v = acc[r] + bias;
      v = v > 0.f ? v : 0.2f * v;                 // LeakyReLU(0.2)
      ldsZ2[(r + mb) * LDA2 + n] = (_Float16)v;
    }
  }
  __syncthreads();
  // GEMM 16x64 @ 64x128 -> 16x128 on all 8 waves
  {
    v8f acc = {};
#pragma unroll
    for (int k0 = 0; k0 < HIDD; k0 += 32) {
      v16h a = load_frag16(ldsZ2, LDA2, k0, lane);
      v16h b = load_frag16(Wt2 + wave * 16 * HIDD, HIDD, k0, lane);
      acc = wmma_f16(a, b, acc);
    }
    int n = wave * 16 + (lane & 15);
    int mb = (lane >> 4) << 3;
    float bias = b2[n];
#pragma unroll
    for (int r = 0; r < 8; ++r) {
      float v = acc[r] + bias;
      ldsNW[(r + mb) * FDIM + n] = 1.f / (1.f + __expf(-v));  // sigmoid
    }
  }
  __syncthreads();
}

// ---------------------------------------------------------------------------
// Weight convert: W[K][N] f32 row-major -> Wt[N][ldt] f16 (transposed,
// zero-padded in K to ldt).  B-fragment friendly layout.
// ---------------------------------------------------------------------------
__global__ void convt_kernel(const float* __restrict__ W,
                             _Float16* __restrict__ Wt,
                             int K, int Nn, int ldt) {
  int i = blockIdx.x * blockDim.x + threadIdx.x;
  int total = Nn * ldt;
  if (i < total) {
    int n = i / ldt, k = i % ldt;
    Wt[i] = (k < K) ? (_Float16)W[(size_t)k * Nn + n] : (_Float16)0.f;
  }
}

// ---------------------------------------------------------------------------
// Encoder + learner0: h0 = [table[idx], x@W_ohe+b_ohe] @ W_enc + b_enc
// h = h0*nw ; codebank = h0*nw*nw.  One block = 16 nodes.
// ---------------------------------------------------------------------------
__global__ __launch_bounds__(256) void encode_kernel(
    const float* __restrict__ x, const int* __restrict__ node_index,
    const float* __restrict__ table,
    const float* __restrict__ W_ohe, const float* __restrict__ b_ohe,
    const _Float16* __restrict__ Wenc, const float* __restrict__ b_enc,
    const _Float16* __restrict__ Wt1, const float* __restrict__ b1,
    const _Float16* __restrict__ Wt2, const float* __restrict__ b2,
    float* __restrict__ h, float* __restrict__ cb, int n_nodes) {
  __shared__ _Float16 ldsA[16 * LDAE];
  __shared__ _Float16 ldsZ[16 * LDA];
  __shared__ _Float16 ldsZ2[16 * LDA2];
  __shared__ float    ldsH[16 * FDIM];
  __shared__ float    ldsNW[16 * FDIM];
  int t = threadIdx.x, lane = t & 31, wave = t >> 5;
  int row0 = blockIdx.x * 16;

  for (int i = t; i < 16 * LDAE; i += 256) ldsA[i] = (_Float16)0.f;
  __syncthreads();
  if (t < 128) {                         // build nf tile: 16 rows x 16 K
    int r = t >> 3, j = t & 7;
    int g = row0 + r; if (g >= n_nodes) g = n_nodes - 1;
    ldsA[r * LDAE + j] = (_Float16)table[(size_t)node_index[g] * 8 + j];
    float o = b_ohe[j];
#pragma unroll
    for (int q = 0; q < 8; ++q) o += x[(size_t)g * 8 + q] * W_ohe[q * 8 + j];
    ldsA[r * LDAE + 8 + j] = (_Float16)o;
  }
  __syncthreads();
  {                                      // h0 = nf @ W_enc (K padded to 32)
    v16h a = load_frag16(ldsA, LDAE, 0, lane);
    v16h b = load_frag16(Wenc + wave * 16 * 32, 32, 0, lane);
    v8f acc = {};
    acc = wmma_f16(a, b, acc);
    int n = wave * 16 + (lane & 15);
    int mb = (lane >> 4) << 3;
    float bias = b_enc[n];
#pragma unroll
    for (int r = 0; r < 8; ++r) {
      float v = acc[r] + bias;
      ldsH[(r + mb) * FDIM + n] = v;
      ldsZ[(r + mb) * LDA + n] = (_Float16)v;   // learner input z = h0
    }
  }
  __syncthreads();
  learner_block(ldsZ, ldsZ2, ldsNW, Wt1, b1, Wt2, b2, lane, wave);
  {                                      // combine + store (8 contiguous cols/thread)
    int r = t >> 4, cc = (t & 15) * 8;
    int g = row0 + r;
    if (g < n_nodes) {
#pragma unroll
      for (int u = 0; u < 8; ++u) {
        int c = cc + u;
        float h0 = ldsH[r * FDIM + c], nw = ldsNW[r * FDIM + c];
        float hn = h0 * nw;
        h[(size_t)g * FDIM + c]  = hn;
        cb[(size_t)g * FDIM + c] = hn * nw;
      }
    }
  }
}

// ---------------------------------------------------------------------------
// Segment-softmax scratch init
// ---------------------------------------------------------------------------
__global__ void init_kernel(int* __restrict__ mx, float* __restrict__ den,
                            float* __restrict__ num, int total) {
  int i = blockIdx.x * blockDim.x + threadIdx.x;
  if (i < total) { mx[i] = (int)0x80000000; den[i] = 0.f; num[i] = 0.f; }
}

// Ordered-int encoding for float atomic max (monotone map)
__device__ __forceinline__ int f2key(float f) {
  int b = __float_as_int(f);
  return b >= 0 ? b : (b ^ 0x7fffffff);
}
__device__ __forceinline__ float key2f(int k) {
  return __int_as_float(k >= 0 ? k : (k ^ 0x7fffffff));
}

// ---------------------------------------------------------------------------
// Edge pass 1: per-edge m = relu(h[src] + edge_attr@W_edge + b) + eps,
// s = m*t, atomicMax into mx[dst].  One wave32 per edge, 4 feats/lane.
// edge_emb is recomputed (W_edge in LDS) -> streams only 32B/edge of attrs;
// h (51 MB) is L2-resident on the 192 MB L2.
// ---------------------------------------------------------------------------
__global__ __launch_bounds__(256) void edge_max_kernel(
    const int* __restrict__ ei, const float* __restrict__ ea,
    const float* __restrict__ h, const float* __restrict__ We,
    const float* __restrict__ be, const float* __restrict__ tptr, int l,
    int n_edges, int* __restrict__ mx) {
  __shared__ float sW[8 * FDIM];
  __shared__ float sB[FDIM];
  for (int i = threadIdx.x; i < 8 * FDIM; i += 256) sW[i] = We[i];
  if (threadIdx.x < FDIM) sB[threadIdx.x] = be[threadIdx.x];
  __syncthreads();
  int e = (blockIdx.x * 256 + threadIdx.x) >> 5;
  int lane = threadIdx.x & 31;
  if (e >= n_edges) return;
  int src = ei[e], dst = ei[n_edges + e];
  float tt = tptr[l];
  float a[8];
#pragma unroll
  for (int j = 0; j < 8; ++j) a[j] = ea[(size_t)e * 8 + j];
  int f0 = lane * 4;
  float4 hv = *reinterpret_cast<const float4*>(h + (size_t)src * FDIM + f0);
  float hvv[4] = {hv.x, hv.y, hv.z, hv.w};
#pragma unroll
  for (int c = 0; c < 4; ++c) {
    int f = f0 + c;
    float ee = sB[f];
#pragma unroll
    for (int j = 0; j < 8; ++j) ee += a[j] * sW[j * FDIM + f];
    float m = fmaxf(hvv[c] + ee, 0.f) + 1e-7f;
    atomicMax(mx + (size_t)dst * FDIM + f, f2key(m * tt));
  }
}

// ---------------------------------------------------------------------------
// Edge pass 2: e = exp(s - mx[dst]); denom += e; numer += e*m.
// ---------------------------------------------------------------------------
__global__ __launch_bounds__(256) void edge_sum_kernel(
    const int* __restrict__ ei, const float* __restrict__ ea,
    const float* __restrict__ h, const float* __restrict__ We,
    const float* __restrict__ be, const float* __restrict__ tptr, int l,
    int n_edges, const int* __restrict__ mx,
    float* __restrict__ den, float* __restrict__ num) {
  __shared__ float sW[8 * FDIM];
  __shared__ float sB[FDIM];
  for (int i = threadIdx.x; i < 8 * FDIM; i += 256) sW[i] = We[i];
  if (threadIdx.x < FDIM) sB[threadIdx.x] = be[threadIdx.x];
  __syncthreads();
  int e = (blockIdx.x * 256 + threadIdx.x) >> 5;
  int lane = threadIdx.x & 31;
  if (e >= n_edges) return;
  int src = ei[e], dst = ei[n_edges + e];
  float tt = tptr[l];
  float a[8];
#pragma unroll
  for (int j = 0; j < 8; ++j) a[j] = ea[(size_t)e * 8 + j];
  int f0 = lane * 4;
  float4 hv = *reinterpret_cast<const float4*>(h + (size_t)src * FDIM + f0);
  float hvv[4] = {hv.x, hv.y, hv.z, hv.w};
#pragma unroll
  for (int c = 0; c < 4; ++c) {
    int f = f0 + c;
    float ee = sB[f];
#pragma unroll
    for (int j = 0; j < 8; ++j) ee += a[j] * sW[j * FDIM + f];
    float m = fmaxf(hvv[c] + ee, 0.f) + 1e-7f;
    float s = m * tt;
    size_t idx = (size_t)dst * FDIM + f;
    float ev = __expf(s - key2f(mx[idx]));
    atomicAdd(den + idx, ev);
    atomicAdd(num + idx, ev * m);
  }
}

// ---------------------------------------------------------------------------
// Fused GENConv layer: agg = numer/(denom+1e-16);
// h1 = (h+agg)@gcn_W + b ; h = relu(LN(h1)) ; nw = learner(h+cb);
// h = h*nw + cb*(1-nw) ; cb += h*nw.  One block = 16 nodes, 8 wave32.
// gcn weights are async-staged into LDS (ASYNCcnt) overlapping the A-tile
// staging; B fragments then come from LDS (ds_load_b128, padded rows).
// ---------------------------------------------------------------------------
__global__ __launch_bounds__(256) void layer_kernel(
    float* __restrict__ h, float* __restrict__ cb,
    const float* __restrict__ num, const float* __restrict__ den,
    const _Float16* __restrict__ Wg, const float* __restrict__ bg,
    const float* __restrict__ lng, const float* __restrict__ lnb,
    const _Float16* __restrict__ Wt1, const float* __restrict__ b1,
    const _Float16* __restrict__ Wt2, const float* __restrict__ b2,
    int n_nodes) {
  __shared__ _Float16 ldsW[FDIM * LDA];   // 128x128 gcn weight tile (padded)
  __shared__ _Float16 ldsA[16 * LDA];
  __shared__ _Float16 ldsZ2[16 * LDA2];
  __shared__ float ldsC[16 * FDIM];
  __shared__ float ldsH[16 * FDIM];
  __shared__ float ldsCB[16 * FDIM];
  __shared__ float ldsNW[16 * FDIM];
  __shared__ float redS[16 * 8], redQ[16 * 8];
  __shared__ float muS[16], rsS[16];
  int t = threadIdx.x, lane = t & 31, wave = t >> 5;
  int row0 = blockIdx.x * 16;

  // Kick off async weight stage: 128 rows x 16 chunks of 16B each.
#pragma unroll
  for (int j = 0; j < 8; ++j) {
    int idx = t + 256 * j;               // 2048 chunks
    int n = idx >> 4, i = idx & 15;
    async_copy_b128(Wg + n * FDIM + i * 8, &ldsW[n * LDA + i * 8]);
  }

  // Stage A = f16(h + agg) (8 contiguous cols/thread -> half8 DS store),
  // stash codebank.  Overlaps with the async weight DMA above.
  {
    int r = t >> 4, cc = (t & 15) * 8;
    int g = row0 + r; if (g >= n_nodes) g = n_nodes - 1;
    size_t gi = (size_t)g * FDIM + cc;
    half8 av;
#pragma unroll
    for (int u = 0; u < 8; ++u) {
      float agg = num[gi + u] / (den[gi + u] + 1e-16f);
      av[u] = (_Float16)(h[gi + u] + agg);
      ldsCB[r * FDIM + cc + u] = cb[gi + u];
    }
    *reinterpret_cast<half8*>(&ldsA[r * LDA + cc]) = av;
  }
  wait_async0();
  __syncthreads();
  // Backbone GEMM 16x128 @ 128x128 (B from LDS)
  {
    v8f acc = {};
#pragma unroll
    for (int k0 = 0; k0 < FDIM; k0 += 32) {
      v16h a = load_frag16(ldsA, LDA, k0, lane);
      v16h b = load_frag16(ldsW + wave * 16 * LDA, LDA, k0, lane);
      acc = wmma_f16(a, b, acc);
    }
    int n = wave * 16 + (lane & 15);
    int mb = (lane >> 4) << 3;
    float bias = bg[n];
#pragma unroll
    for (int r = 0; r < 8; ++r) ldsC[(r + mb) * FDIM + n] = acc[r] + bias;
  }
  __syncthreads();
  // LayerNorm row stats (16 rows x 8 partials)
  if (t < 128) {
    int r = t >> 3, p = t & 7;
    float s = 0.f, q = 0.f;
    for (int j = 0; j < 16; ++j) {
      float v = ldsC[r * FDIM + p * 16 + j];
      s += v; q += v * v;
    }
    redS[r * 8 + p] = s; redQ[r * 8 + p] = q;
  }
  __syncthreads();
  if (t < 16) {
    float s = 0.f, q = 0.f;
    for (int p = 0; p < 8; ++p) { s += redS[t * 8 + p]; q += redQ[t * 8 + p]; }
    float mu = s * (1.f / FDIM);
    muS[t] = mu;
    rsS[t] = rsqrtf(q * (1.f / FDIM) - mu * mu + 1e-5f);
  }
  __syncthreads();
  // LN + relu -> ldsH ; learner input z = h_new + cb -> ldsA (half8 store)
  {
    int r = t >> 4, cc = (t & 15) * 8;
    float mu = muS[r], rs = rsS[r];
    half8 zv;
#pragma unroll
    for (int u = 0; u < 8; ++u) {
      int c = cc + u;
      float v = lng[c] * (ldsC[r * FDIM + c] - mu) * rs + lnb[c];
      v = fmaxf(v, 0.f);
      ldsH[r * FDIM + c] = v;
      zv[u] = (_Float16)(v + ldsCB[r * FDIM + c]);
    }
    *reinterpret_cast<half8*>(&ldsA[r * LDA + cc]) = zv;
  }
  __syncthreads();
  learner_block(ldsA, ldsZ2, ldsNW, Wt1, b1, Wt2, b2, lane, wave);
  // Highway combine + codebank update (contiguous 32B global stores)
  {
    int r = t >> 4, cc = (t & 15) * 8;
    int g = row0 + r;
    if (g < n_nodes) {
#pragma unroll
      for (int u = 0; u < 8; ++u) {
        int c = cc + u;
        float hn = ldsH[r * FDIM + c], nw = ldsNW[r * FDIM + c];
        float c0 = ldsCB[r * FDIM + c];
        float hf = hn * nw;
        h[(size_t)g * FDIM + c]  = hf + c0 * (1.f - nw);
        cb[(size_t)g * FDIM + c] = c0 + hf;
      }
    }
  }
}

// ---------------------------------------------------------------------------
// Head: out = codebank @ W_pred + b_pred   (128 -> 112, 7 column tiles)
// W_pred async-staged into LDS.
// ---------------------------------------------------------------------------
__global__ __launch_bounds__(256) void pred_kernel(
    const float* __restrict__ cb, const _Float16* __restrict__ Wp,
    const float* __restrict__ bp, float* __restrict__ out,
    int n_nodes, int tasks) {
  __shared__ _Float16 ldsW[112 * LDA];
  __shared__ _Float16 ldsA[16 * LDA];
  int t = threadIdx.x, lane = t & 31, wave = t >> 5;
  int row0 = blockIdx.x * 16;
  // Async stage W_pred: tasks rows x 16 chunks of 16B.
  for (int idx = t; idx < tasks * 16; idx += 256) {
    int n = idx >> 4, i = idx & 15;
    async_copy_b128(Wp + n * FDIM + i * 8, &ldsW[n * LDA + i * 8]);
  }
  {
    int r = t >> 4, cc = (t & 15) * 8;
    int g = row0 + r; if (g >= n_nodes) g = n_nodes - 1;
    half8 av;
#pragma unroll
    for (int u = 0; u < 8; ++u)
      av[u] = (_Float16)cb[(size_t)g * FDIM + cc + u];
    *reinterpret_cast<half8*>(&ldsA[r * LDA + cc]) = av;
  }
  wait_async0();
  __syncthreads();
  int ntiles = (tasks + 15) / 16;
  if (wave < ntiles) {
    v8f acc = {};
#pragma unroll
    for (int k0 = 0; k0 < FDIM; k0 += 32) {
      v16h a = load_frag16(ldsA, LDA, k0, lane);
      v16h b = load_frag16(ldsW + wave * 16 * LDA, LDA, k0, lane);
      acc = wmma_f16(a, b, acc);
    }
    int n = wave * 16 + (lane & 15);
    int mb = (lane >> 4) << 3;
    if (n < tasks) {
      float bias = bp[n];
#pragma unroll
      for (int r = 0; r < 8; ++r) {
        int g = row0 + r + mb;
        if (g < n_nodes) out[(size_t)g * tasks + n] = acc[r] + bias;
      }
    }
  }
}

// ---------------------------------------------------------------------------
// Launcher
// ---------------------------------------------------------------------------
extern "C" void kernel_launch(void* const* d_in, const int* in_sizes, int n_in,
                              void* d_out, int out_size, void* d_ws, size_t ws_size,
                              hipStream_t stream) {
  const float* x          = (const float*)d_in[0];
  const int*   node_index = (const int*)d_in[1];
  const int*   edge_index = (const int*)d_in[2];
  const float* edge_attr  = (const float*)d_in[3];
  const float* table      = (const float*)d_in[4];
  const float* W_ohe  = (const float*)d_in[5];
  const float* b_ohe  = (const float*)d_in[6];
  const float* W_enc  = (const float*)d_in[7];
  const float* b_enc  = (const float*)d_in[8];
  const float* W_edge = (const float*)d_in[9];
  const float* b_edge = (const float*)d_in[10];
  const float* lW1    = (const float*)d_in[11];
  const float* lb1    = (const float*)d_in[12];
  const float* lW2    = (const float*)d_in[13];
  const float* lb2    = (const float*)d_in[14];
  const float* gcn_W  = (const float*)d_in[15];
  const float* gcn_b  = (const float*)d_in[16];
  const float* gcn_t  = (const float*)d_in[17];
  const float* ln_g   = (const float*)d_in[18];
  const float* ln_b   = (const float*)d_in[19];
  const float* W_pred = (const float*)d_in[20];
  const float* b_pred = (const float*)d_in[21];

  const int N = in_sizes[1];
  const int E = in_sizes[2] / 2;
  const int L = in_sizes[17];
  const int TASKS = in_sizes[21];

  // Workspace carve-up (256B aligned chunks)
  size_t off = 0;
  auto carve = [&](size_t bytes) -> char* {
    char* p = (char*)d_ws + off;
    off += (bytes + 255) & ~(size_t)255;
    return p;
  };
  float* h   = (float*)carve((size_t)N * FDIM * 4);
  float* cb  = (float*)carve((size_t)N * FDIM * 4);
  int*   mx  = (int*)  carve((size_t)N * FDIM * 4);
  float* den = (float*)carve((size_t)N * FDIM * 4);
  float* num = (float*)carve((size_t)N * FDIM * 4);
  _Float16* WgT  = (_Float16*)carve((size_t)L * FDIM * FDIM * 2);
  _Float16* W1T  = (_Float16*)carve((size_t)(L + 1) * HIDD * FDIM * 2);
  _Float16* W2T  = (_Float16*)carve((size_t)(L + 1) * FDIM * HIDD * 2);
  _Float16* WeT  = (_Float16*)carve((size_t)FDIM * 32 * 2);
  _Float16* WpT  = (_Float16*)carve((size_t)TASKS * FDIM * 2);

  const int NB  = (N + 15) / 16;      // node-tile blocks
  const int EB  = (E + 7) / 8;        // 8 edges (waves) per block
  const int IB  = (N * FDIM + 255) / 256;

  // --- Weight conversion/transposition to f16 B-layout ---
  convt_kernel<<<(FDIM * 32 + 255) / 256, 256, 0, stream>>>(W_enc, WeT, 16, FDIM, 32);
  for (int l = 0; l < L; ++l)
    convt_kernel<<<(FDIM * FDIM + 255) / 256, 256, 0, stream>>>(
        gcn_W + (size_t)l * FDIM * FDIM, WgT + (size_t)l * FDIM * FDIM, FDIM, FDIM, FDIM);
  for (int li = 0; li <= L; ++li) {
    convt_kernel<<<(HIDD * FDIM + 255) / 256, 256, 0, stream>>>(
        lW1 + (size_t)li * FDIM * HIDD, W1T + (size_t)li * HIDD * FDIM, FDIM, HIDD, FDIM);
    convt_kernel<<<(FDIM * HIDD + 255) / 256, 256, 0, stream>>>(
        lW2 + (size_t)li * HIDD * FDIM, W2T + (size_t)li * FDIM * HIDD, HIDD, FDIM, HIDD);
  }
  convt_kernel<<<(TASKS * FDIM + 255) / 256, 256, 0, stream>>>(W_pred, WpT, FDIM, TASKS, FDIM);

  // --- Encoder + learner0 ---
  encode_kernel<<<NB, 256, 0, stream>>>(x, node_index, table, W_ohe, b_ohe,
                                        WeT, b_enc, W1T, lb1, W2T, lb2,
                                        h, cb, N);

  // --- GENConv layers ---
  for (int l = 0; l < L; ++l) {
    init_kernel<<<IB, 256, 0, stream>>>(mx, den, num, N * FDIM);
    edge_max_kernel<<<EB, 256, 0, stream>>>(edge_index, edge_attr, h,
                                            W_edge, b_edge, gcn_t, l, E, mx);
    edge_sum_kernel<<<EB, 256, 0, stream>>>(edge_index, edge_attr, h,
                                            W_edge, b_edge, gcn_t, l, E,
                                            mx, den, num);
    layer_kernel<<<NB, 256, 0, stream>>>(
        h, cb, num, den,
        WgT + (size_t)l * FDIM * FDIM, gcn_b + (size_t)l * FDIM,
        ln_g + (size_t)l * FDIM, ln_b + (size_t)l * FDIM,
        W1T + (size_t)(l + 1) * HIDD * FDIM, lb1 + (size_t)(l + 1) * HIDD,
        W2T + (size_t)(l + 1) * FDIM * HIDD, lb2 + (size_t)(l + 1) * FDIM, N);
  }

  // --- Prediction head ---
  pred_kernel<<<NB, 256, 0, stream>>>(cb, WpT, b_pred, (float*)d_out, N, TASKS);
}
